// Attention_6150393168242
// MI455X (gfx1250) — matrix operand order repaired and checked by reading, not compiled, not run on hardware
//
#include <hip/hip_runtime.h>

// Problem constants (match reference: B=4, N=2048, D=1024)
#define BATCH  4
#define SEQ    2048
#define DMODEL 1024

typedef __attribute__((ext_vector_type(16))) __bf16 v16bf;
typedef __attribute__((ext_vector_type(8)))  __bf16 v8bf;
typedef __attribute__((ext_vector_type(8)))  float  v8f;

// fp32 -> bf16, round-to-nearest-even
static __device__ __forceinline__ __bf16 f2bf(float f) {
  unsigned u = __builtin_bit_cast(unsigned, f);
  unsigned r = u + 0x7FFFu + ((u >> 16) & 1u);
  unsigned short h = (unsigned short)(r >> 16);
  return __builtin_bit_cast(__bf16, h);
}

static __device__ __forceinline__ void cstore(float* p, float v)  { *p = v; }
static __device__ __forceinline__ void cstore(__bf16* p, float v) { *p = f2bf(v); }

__global__ void __launch_bounds__(256)
f32_to_bf16(const float* __restrict__ in, __bf16* __restrict__ out, size_t n) {
  size_t i = (size_t)blockIdx.x * blockDim.x + threadIdx.x;
  size_t stride = (size_t)gridDim.x * blockDim.x;
  for (; i < n; i += stride) out[i] = f2bf(in[i]);
}

// ---------------------------------------------------------------------------
// Generic NT GEMM on bf16 with f32 WMMA accumulation:
//   C[b][i][j] = alpha * sum_k A[b][i][k] * Bm[b][j][k]
//
// One wave32 computes a 32(M) x 64(N) tile: 2 A-fragments x 4 B-fragments
// -> 8 v_wmma_f32_16x16x32_bf16 per 12 global_load_b128 per 32-deep K step.
// The K loop is a 2-deep software pipeline (ping-pong fragment buffers) so
// the loads of stage i+1 are in flight while the WMMAs of stage i execute,
// letting the compiler issue partial s_wait_loadcnt instead of draining.
//
// Fragment packing per CDNA5 ISA 7.12.2 (bf16 16x16x32):
//  A: lane m<16 -> row m, K {0..7,16..23}; lane m+16 -> K {8..15,24..31}
//  B: lane n<16 -> col n (== NT-operand row), K {0..7,16..23}; lane n+16 -> K {8..15,24..31}
//  C/D: VGPR i -> row i + 8*half, lane r -> col
// ---------------------------------------------------------------------------
union Frag { v16bf v; v8bf h[2]; };
struct Frags { Frag a[2]; Frag b[4]; };

template <typename OutT>
__global__ void __launch_bounds__(128)
gemm_nt_bf16(const __bf16* __restrict__ A, const __bf16* __restrict__ Bm,
             OutT* __restrict__ C, int Kd, int lda, int ldb, int ldc,
             long long sA, long long sB, long long sC, float alpha) {
  const int lane = threadIdx.x & 31;
  const int wave = threadIdx.x >> 5;
  const int half = (lane >> 4) & 1;
  const int r    = lane & 15;

  const __bf16* Ab = A  + (long long)blockIdx.z * sA;
  const __bf16* Bb = Bm + (long long)blockIdx.z * sB;
  OutT*         Cb = C  + (long long)blockIdx.z * sC;

  const int tileM = blockIdx.y * 128 + wave * 32;   // 32 rows per wave
  const int tileN = blockIdx.x * 64;                // 64 cols per wave

  const __bf16* arow[2];
#pragma unroll
  for (int m = 0; m < 2; ++m)
    arow[m] = Ab + (size_t)(tileM + m * 16 + r) * lda;
  const __bf16* brow[4];
#pragma unroll
  for (int t = 0; t < 4; ++t)
    brow[t] = Bb + (size_t)(tileN + t * 16 + r) * ldb;

  auto load_frags = [&](Frags& f, int k0) {
#pragma unroll
    for (int m = 0; m < 2; ++m) {
      f.a[m].h[0] = *(const v8bf*)(arow[m] + k0 + half * 8);
      f.a[m].h[1] = *(const v8bf*)(arow[m] + k0 + 16 + half * 8);
    }
#pragma unroll
    for (int t = 0; t < 4; ++t) {
      f.b[t].h[0] = *(const v8bf*)(brow[t] + k0 + half * 8);
      f.b[t].h[1] = *(const v8bf*)(brow[t] + k0 + 16 + half * 8);
    }
  };

  v8f acc[2][4] = {};
  auto compute = [&](const Frags& f) {
#pragma unroll
    for (int t = 0; t < 4; ++t) {
      acc[0][t] = __builtin_amdgcn_wmma_f32_16x16x32_bf16(
          false, f.a[0].v, false, f.b[t].v, (short)0, acc[0][t], false, false);
      acc[1][t] = __builtin_amdgcn_wmma_f32_16x16x32_bf16(
          false, f.a[1].v, false, f.b[t].v, (short)0, acc[1][t], false, false);
    }
  };

  // 2-deep pipelined K loop (Kd is a multiple of 64 for all uses here)
  Frags f0, f1;
  load_frags(f0, 0);
  for (int k0 = 0; k0 < Kd; k0 += 64) {
    load_frags(f1, k0 + 32);
    compute(f0);
    if (k0 + 64 < Kd) load_frags(f0, k0 + 64);
    compute(f1);
  }

#pragma unroll
  for (int m = 0; m < 2; ++m) {
#pragma unroll
    for (int t = 0; t < 4; ++t) {
#pragma unroll
      for (int i = 0; i < 8; ++i) {
        size_t row = (size_t)(tileM + m * 16 + half * 8 + i);
        size_t col = (size_t)(tileN + t * 16 + r);
        cstore(&Cb[row * ldc + col], acc[m][t][i] * alpha);
      }
    }
  }
}

// ---------------------------------------------------------------------------
// In-place row softmax over SEQ fp32 logits; writes bf16 probabilities over
// the front half of the same row (row stays at a 2*SEQ bf16-element stride).
// One 256-thread block per row; all reads complete before the first barrier,
// so overwriting the low 4 KB of the 8 KB fp32 row is safe.
// ---------------------------------------------------------------------------
__global__ void __launch_bounds__(256)
softmax_rows(float* __restrict__ S) {
  const int t = threadIdx.x;
  float*  rs = S + (size_t)blockIdx.x * SEQ;
  __bf16* rp = (__bf16*)rs;
  __shared__ float red[256];

  float v[SEQ / 256];
  float mx = -3.402823466e38f;
#pragma unroll
  for (int j = 0; j < SEQ / 256; ++j) {
    v[j] = rs[t + j * 256];
    mx = fmaxf(mx, v[j]);
  }
  red[t] = mx; __syncthreads();
  for (int s = 128; s > 0; s >>= 1) {
    if (t < s) red[t] = fmaxf(red[t], red[t + s]);
    __syncthreads();
  }
  mx = red[0]; __syncthreads();

  float sum = 0.f;
#pragma unroll
  for (int j = 0; j < SEQ / 256; ++j) {
    v[j] = __expf(v[j] - mx);
    sum += v[j];
  }
  red[t] = sum; __syncthreads();
  for (int s = 128; s > 0; s >>= 1) {
    if (t < s) red[t] += red[t + s];
    __syncthreads();
  }
  const float inv = 1.0f / red[0];
#pragma unroll
  for (int j = 0; j < SEQ / 256; ++j)
    rp[t + j * 256] = f2bf(v[j] * inv);
}

// ---------------------------------------------------------------------------
extern "C" void kernel_launch(void* const* d_in, const int* in_sizes, int n_in,
                              void* d_out, int out_size, void* d_ws, size_t ws_size,
                              hipStream_t stream) {
  (void)in_sizes; (void)n_in; (void)out_size; (void)ws_size;
  const float* x  = (const float*)d_in[0];
  const float* Wq = (const float*)d_in[1];
  const float* Wk = (const float*)d_in[2];
  const float* Wv = (const float*)d_in[3];
  float* out = (float*)d_out;

  const size_t nxd = (size_t)BATCH * SEQ * DMODEL;   // 8M elems
  const size_t ndd = (size_t)DMODEL * DMODEL;        // 1M elems

  // Workspace layout (all L2-resident: ~134 MB < 192 MB L2)
  __bf16* xb   = (__bf16*)d_ws;      // x in bf16            [B*N, D]
  __bf16* wqb  = xb  + nxd;          // Wq bf16              [D, D]
  __bf16* wkb  = wqb + ndd;
  __bf16* wvb  = wkb + ndd;
  __bf16* Qb   = wvb + ndd;          // Q bf16               [B*N, D]
  __bf16* Kbuf = Qb  + nxd;          // K bf16               [B*N, D]
  __bf16* Vt   = Kbuf + nxd;         // V^T bf16             [B][D][N]
  float*  Sf   = (float*)(Vt + nxd); // scores fp32 / P bf16 [B][N][N]

  const dim3 blk(128);
  const float scale = 0.03125f;      // 1/sqrt(1024)

  // 0) fp32 -> bf16 conversions
  f32_to_bf16<<<1024, 256, 0, stream>>>(x,  xb,  nxd);
  f32_to_bf16<<<256,  256, 0, stream>>>(Wq, wqb, ndd);
  f32_to_bf16<<<256,  256, 0, stream>>>(Wk, wkb, ndd);
  f32_to_bf16<<<256,  256, 0, stream>>>(Wv, wvb, ndd);

  // 1) Q = x @ Wq^T   (single GEMM over fused B*N rows)
  gemm_nt_bf16<__bf16><<<dim3(DMODEL / 64, (BATCH * SEQ) / 128, 1), blk, 0, stream>>>(
      xb, wqb, Qb, DMODEL, DMODEL, DMODEL, DMODEL, 0LL, 0LL, 0LL, 1.0f);
  // 2) K = x @ Wk^T
  gemm_nt_bf16<__bf16><<<dim3(DMODEL / 64, (BATCH * SEQ) / 128, 1), blk, 0, stream>>>(
      xb, wkb, Kbuf, DMODEL, DMODEL, DMODEL, DMODEL, 0LL, 0LL, 0LL, 1.0f);
  // 3) V^T[b] = Wv @ x[b]^T   (batched; A stride 0 shares Wv)
  gemm_nt_bf16<__bf16><<<dim3(SEQ / 64, DMODEL / 128, BATCH), blk, 0, stream>>>(
      wvb, xb, Vt, DMODEL, DMODEL, DMODEL, SEQ,
      0LL, (long long)SEQ * DMODEL, (long long)DMODEL * SEQ, 1.0f);
  // 4) S[b] = scale * Q[b] @ K[b]^T   (fp32 out for softmax)
  gemm_nt_bf16<float><<<dim3(SEQ / 64, SEQ / 128, BATCH), blk, 0, stream>>>(
      Qb, Kbuf, Sf, DMODEL, DMODEL, DMODEL, SEQ,
      (long long)SEQ * DMODEL, (long long)SEQ * DMODEL, (long long)SEQ * SEQ, scale);
  // 5) softmax rows in place: fp32 S -> bf16 P (row stride 2*SEQ bf16 elems)
  softmax_rows<<<BATCH * SEQ, 256, 0, stream>>>(Sf);
  // 6) out[b] = P[b] @ V[b] = P[b] @ (V^T[b])^T
  gemm_nt_bf16<float><<<dim3(DMODEL / 64, SEQ / 128, BATCH), blk, 0, stream>>>(
      (const __bf16*)Sf, Vt, out, SEQ, 2 * SEQ, SEQ, DMODEL,
      2LL * SEQ * SEQ, (long long)DMODEL * SEQ, (long long)SEQ * DMODEL, 1.0f);
}
